// SNLB_46145128628248
// MI455X (gfx1250) — compile-verified, matrix operand order checked by reading, not run on hardware
//
#include <hip/hip_runtime.h>
#include <hip/hip_bf16.h>

#define N_POS 4096
#define C_IN  256
#define C_I   128

#ifndef __has_builtin
#define __has_builtin(x) 0
#endif
#if defined(__AMDGCN__) && __has_builtin(__builtin_amdgcn_global_load_async_to_lds_b128) && \
    __has_builtin(__builtin_amdgcn_s_wait_asynccnt)
#define HAVE_ASYNC 1
#else
#define HAVE_ASYNC 0
#endif

typedef __attribute__((ext_vector_type(16))) __bf16 v16bf;
typedef __attribute__((ext_vector_type(8)))  __bf16 v8bf;
typedef __attribute__((ext_vector_type(8)))  float  v8f;
typedef __attribute__((__vector_size__(16))) int v4i_t;  // matches async-builtin param

union V16 { v16bf v; uint4 q[2]; unsigned short u[16]; };
union V8  { v8bf  v; uint4 q;    unsigned short u[8];  };

__device__ __forceinline__ unsigned short f2bfu(float f) {
  unsigned u = __builtin_bit_cast(unsigned, f);
  return (unsigned short)((u + 0x7FFFu + ((u >> 16) & 1u)) >> 16);  // RNE
}
__device__ __forceinline__ __bf16 f2bf(float f) {
  unsigned short s = f2bfu(f);
  return __builtin_bit_cast(__bf16, s);
}
__device__ __forceinline__ v8f zero8() {
  v8f z = {0.f, 0.f, 0.f, 0.f, 0.f, 0.f, 0.f, 0.f};
  return z;
}

// 16-bit fragment load, matching the ISA 16x32 A layout (B assumed symmetric):
// lane m (0-15): K = k0+0..7 (elems 0-7), k0+16..23 (elems 8-15)
// lane m+16   : K = k0+8..15,            k0+24..31
__device__ __forceinline__ v16bf load_frag(const __bf16* __restrict__ p, int ld, int row0, int k0) {
  const int l  = threadIdx.x & 31;
  const int r  = row0 + (l & 15);
  const int kb = k0 + ((l & 16) ? 8 : 0);
  const __bf16* q = p + (size_t)r * (size_t)ld + kb;
  V16 t;
  t.q[0] = *(const uint4*)(q);
  t.q[1] = *(const uint4*)(q + 16);
  return t.v;
}

__device__ __forceinline__ v8f wmma_bf16(v16bf a, v16bf b, v8f c) {
  return __builtin_amdgcn_wmma_f32_16x16x32_bf16(false, a, false, b, (short)0, c, false, false);
}

// 16-byte global -> LDS copy (async on CDNA5, sync fallback)
__device__ __forceinline__ void copy16(const __bf16* g, __bf16* l) {
#if HAVE_ASYNC
  __builtin_amdgcn_global_load_async_to_lds_b128(
      (__attribute__((address_space(1))) v4i_t*)g,
      (__attribute__((address_space(3))) v4i_t*)l, 0, 0);
#else
  *(uint4*)l = *(const uint4*)g;
#endif
}
__device__ __forceinline__ void wait_async() {
#if HAVE_ASYNC
  __builtin_amdgcn_s_wait_asynccnt(0);
#endif
}

// ---------------- fp32 -> bf16 convert (weights) ----------------
__global__ void k_cvt(const float* __restrict__ in, __bf16* __restrict__ out, int n) {
  int i = blockIdx.x * blockDim.x + threadIdx.x;
  if (i < n) out[i] = f2bf(in[i]);
}

// ---------------- x [B,C,N] f32 -> xt [B,N,C] bf16 ----------------
__global__ void k_xpose(const float* __restrict__ x, __bf16* __restrict__ xt) {
  __shared__ float tile[64][65];
  const int b  = blockIdx.z;
  const int n0 = blockIdx.x * 64;
  const int c0 = blockIdx.y * 64;
  for (int cc = threadIdx.y; cc < 64; cc += 4)
    tile[cc][threadIdx.x] = x[((size_t)(b * C_IN + c0 + cc)) * N_POS + n0 + threadIdx.x];
  __syncthreads();
  for (int nn = threadIdx.y; nn < 64; nn += 4)
    xt[((size_t)(b * N_POS + n0 + nn)) * C_IN + c0 + threadIdx.x] = f2bf(tile[threadIdx.x][nn]);
}

// ---------------- projections: theta/phi [B,N,CI], g [B,CI,N] ----------------
__global__ void __launch_bounds__(128) k_proj(
    const __bf16* __restrict__ xt,
    const __bf16* __restrict__ wtb, const __bf16* __restrict__ wpb, const __bf16* __restrict__ wgb,
    const float* __restrict__ bt, const float* __restrict__ bp, const float* __restrict__ bg,
    __bf16* __restrict__ th, __bf16* __restrict__ ph, __bf16* __restrict__ gch) {
  const int wgid = blockIdx.x * 4 + (threadIdx.x >> 5);
  const int proj = wgid >> 10;            // 0:theta 1:phi 2:g
  const int b    = (wgid >> 8) & 3;
  const int n0   = (wgid & 255) * 16;
  const int l    = threadIdx.x & 31;
  const int hl   = (l >> 4) & 1;

  const __bf16* wb   = (proj == 0) ? wtb : (proj == 1) ? wpb : wgb;
  const float*  bias = (proj == 0) ? bt  : (proj == 1) ? bp  : bg;
  const __bf16* xb   = xt + (size_t)b * N_POS * C_IN;

  v8f acc[8];
  #pragma unroll
  for (int t = 0; t < 8; ++t) acc[t] = zero8();

  for (int ks = 0; ks < 8; ++ks) {
    v16bf a = load_frag(xb, C_IN, n0, ks * 32);
    #pragma unroll
    for (int t = 0; t < 8; ++t) {
      v16bf w = load_frag(wb, C_IN, t * 16, ks * 32);
      acc[t] = wmma_bf16(a, w, acc[t]);
    }
  }

  if (proj < 2) {  // position-major [N, CI]
    __bf16* out = (proj == 0) ? th : ph;
    #pragma unroll
    for (int t = 0; t < 8; ++t) {
      const int ci = t * 16 + (l & 15);
      const float bia = bias[ci];
      #pragma unroll
      for (int r = 0; r < 8; ++r) {
        const int n = n0 + r + 8 * hl;
        out[((size_t)(b * N_POS + n)) * C_I + ci] = f2bf(acc[t][r] + bia);
      }
    }
  } else {         // channel-major [CI, N]: 16B contiguous stores per lane
    #pragma unroll
    for (int t = 0; t < 8; ++t) {
      const int ci = t * 16 + (l & 15);
      const float bia = bias[ci];
      V8 o;
      #pragma unroll
      for (int r = 0; r < 8; ++r) o.u[r] = f2bfu(acc[t][r] + bia);
      *(v8bf*)(gch + ((size_t)(b * C_I + ci)) * N_POS + n0 + 8 * hl) = o.v;
    }
  }
}

// ---------------- fused flash attention: y = softmax(theta.phiT).g ----------------
// Block = 4 waves sharing one batch's 32-m phi/g tile via double-buffered LDS,
// staged with async global->LDS copies. Each wave owns 16 n-rows.
__global__ void __launch_bounds__(128) k_attn(
    const __bf16* __restrict__ th, const __bf16* __restrict__ ph,
    const __bf16* __restrict__ gch, __bf16* __restrict__ yb) {
  __shared__ __bf16 sphi[2][32 * C_I];   // 2 x 8KB: phi rows m0..m0+31, ld=128
  __shared__ __bf16 sg[2][C_I * 32];     // 2 x 8KB: g[ci][m-m0], ld=32

  const int bid = blockIdx.x;            // 256 blocks: b * 64 + nblk
  const int b   = bid >> 6;
  const int n0  = ((bid & 63) * 4 + (threadIdx.x >> 5)) * 16;  // this wave's rows
  const int l   = threadIdx.x & 31;
  const int hb  = (l & 16) ? 8 : 0;
  const int tid = threadIdx.x;

  const __bf16* tb = th  + (size_t)b * N_POS * C_I;
  const __bf16* pb = ph  + (size_t)b * N_POS * C_I;
  const __bf16* gb = gch + (size_t)b * C_I * N_POS;

  v16bf tf[4];  // theta B-fragments for this wave's 16 rows, K = CI
  #pragma unroll
  for (int ks = 0; ks < 4; ++ks) tf[ks] = load_frag(tb, C_I, n0, ks * 32);

  v8f acc[8];   // y accumulators: 16 n x 128 ci
  #pragma unroll
  for (int t = 0; t < 8; ++t) acc[t] = zero8();
  float mprev = -3.0e38f;
  float lsum  = 0.0f;

  // cooperative stage of one 32-m tile (phi 8KB + g 8KB), 16B per thread-chunk
  auto stage = [&](int buf, int m0) {
    #pragma unroll
    for (int p = 0; p < 4; ++p) {            // phi: 512 chunks of 16B
      const int q = p * 128 + tid;
      const int row = q >> 4, ck = (q & 15) * 8;
      copy16(pb + (size_t)(m0 + row) * C_I + ck, &sphi[buf][row * C_I + ck]);
    }
    #pragma unroll
    for (int p = 0; p < 4; ++p) {            // g: 512 chunks of 16B
      const int q = p * 128 + tid;
      const int row = q >> 2, ck = (q & 3) * 8;
      copy16(gb + (size_t)row * N_POS + m0 + ck, &sg[buf][row * 32 + ck]);
    }
  };

  stage(0, 0);
  for (int mb = 0; mb < N_POS / 32; ++mb) {
    wait_async();        // this wave's copies for buf cur are done
    __syncthreads();     // all waves' copies done; all done reading prev buf
    if (mb + 1 < N_POS / 32) stage((mb + 1) & 1, (mb + 1) * 32);

    const __bf16* sp = &sphi[mb & 1][0];
    const __bf16* sgp = &sg[mb & 1][0];

    // S^T tiles from LDS: D-layout (lane=n col, vgpr=m row) == A-layout of P
    v8f st0 = zero8(), st1 = zero8();
    #pragma unroll
    for (int ks = 0; ks < 4; ++ks) {
      v16bf a0 = load_frag(sp, C_I, 0,  ks * 32);
      st0 = wmma_bf16(a0, tf[ks], st0);
      v16bf a1 = load_frag(sp, C_I, 16, ks * 32);
      st1 = wmma_bf16(a1, tf[ks], st1);
    }

    // online softmax: lane pair (l, l^16) holds all 32 m of row n=l&15
    float tmax = -3.0e38f;
    #pragma unroll
    for (int j = 0; j < 8; ++j) { tmax = fmaxf(tmax, st0[j]); tmax = fmaxf(tmax, st1[j]); }
    tmax = fmaxf(tmax, __shfl_xor(tmax, 16, 32));
    const float mnew  = fmaxf(mprev, tmax);
    const float scale = __expf(mprev - mnew);

    V16 pa;
    float ls = 0.0f;
    #pragma unroll
    for (int j = 0; j < 8; ++j) { float e = __expf(st0[j] - mnew); ls += e; pa.u[j]     = f2bfu(e); }
    #pragma unroll
    for (int j = 0; j < 8; ++j) { float e = __expf(st1[j] - mnew); ls += e; pa.u[8 + j] = f2bfu(e); }
    ls += __shfl_xor(ls, 16, 32);
    lsum  = lsum * scale + ls;
    mprev = mnew;

    // rescale y acc (per-row scale broadcast: row n = r + 8*(l>=16))
    #pragma unroll
    for (int r = 0; r < 8; ++r) {
      const float sr = __shfl(scale, r + hb, 32);
      #pragma unroll
      for (int t = 0; t < 8; ++t) acc[t][r] *= sr;
    }

    // y += P (16n x 32m) * g (32m x 16ci) from LDS (ld=32, relative m)
    #pragma unroll
    for (int t = 0; t < 8; ++t) {
      v16bf gf = load_frag(sgp, 32, t * 16, 0);
      acc[t] = wmma_bf16(pa.v, gf, acc[t]);
    }
  }

  const float inv = 1.0f / lsum;
  float invr[8];
  #pragma unroll
  for (int r = 0; r < 8; ++r) invr[r] = __shfl(inv, r + hb, 32);

  const int hl = hb ? 1 : 0;
  #pragma unroll
  for (int t = 0; t < 8; ++t) {
    const int ci = t * 16 + (l & 15);
    #pragma unroll
    for (int r = 0; r < 8; ++r) {
      const int n = n0 + r + 8 * hl;
      yb[((size_t)(b * N_POS + n)) * C_I + ci] = f2bf(acc[t][r] * invr[r]);
    }
  }
}

// ---------------- out = BN(y @ w_outT + b_out) + x ----------------
__global__ void __launch_bounds__(128) k_out(
    const __bf16* __restrict__ yb, const __bf16* __restrict__ wob,
    const float* __restrict__ bo, const float* __restrict__ gamma, const float* __restrict__ beta,
    const float* __restrict__ mean, const float* __restrict__ var,
    const float* __restrict__ x, float* __restrict__ out) {
  const int wgid = blockIdx.x * 4 + (threadIdx.x >> 5);
  const int b   = wgid >> 9;
  const int rem = wgid & 511;
  const int n0  = (rem >> 1) * 16;
  const int c0  = (rem & 1) * 128;
  const int l   = threadIdx.x & 31;
  const int hl  = (l >> 4) & 1;

  const __bf16* ybase = yb + (size_t)b * N_POS * C_I;

  v8f acc[8];
  #pragma unroll
  for (int t = 0; t < 8; ++t) acc[t] = zero8();

  for (int ks = 0; ks < 4; ++ks) {
    v16bf a = load_frag(ybase, C_I, n0, ks * 32);
    #pragma unroll
    for (int t = 0; t < 8; ++t) {
      v16bf w = load_frag(wob, C_I, c0 + t * 16, ks * 32);
      acc[t] = wmma_bf16(a, w, acc[t]);
    }
  }

  #pragma unroll
  for (int t = 0; t < 8; ++t) {
    const int c = c0 + t * 16 + (l & 15);
    const float iv  = gamma[c] * rsqrtf(var[c] + 1e-5f);
    const float add = (bo[c] - mean[c]) * iv + beta[c];
    const size_t base = ((size_t)(b * C_IN + c)) * N_POS + n0 + 8 * hl;
    #pragma unroll
    for (int r = 0; r < 8; ++r)
      out[base + r] = acc[t][r] * iv + add + x[base + r];
  }
}

extern "C" void kernel_launch(void* const* d_in, const int* in_sizes, int n_in,
                              void* d_out, int out_size, void* d_ws, size_t ws_size,
                              hipStream_t stream) {
  const float* x  = (const float*)d_in[0];
  const float* wt = (const float*)d_in[1];
  const float* bt = (const float*)d_in[2];
  const float* wp = (const float*)d_in[3];
  const float* bp = (const float*)d_in[4];
  const float* wg = (const float*)d_in[5];
  const float* bg = (const float*)d_in[6];
  const float* wo = (const float*)d_in[7];
  const float* bo = (const float*)d_in[8];
  const float* gm = (const float*)d_in[9];
  const float* bb = (const float*)d_in[10];
  const float* mn = (const float*)d_in[11];
  const float* vr = (const float*)d_in[12];
  float* out = (float*)d_out;

  char* ws = (char*)d_ws;
  __bf16* wtb = (__bf16*)(ws + 0);         //  64 KB
  __bf16* wpb = (__bf16*)(ws + 65536);
  __bf16* wgb = (__bf16*)(ws + 131072);
  __bf16* wob = (__bf16*)(ws + 196608);
  __bf16* xt  = (__bf16*)(ws + 262144);    // 8 MB  [B,N,C] bf16
  __bf16* th  = (__bf16*)(ws + 8650752);   // 4 MB  [B,N,CI]
  __bf16* ph  = (__bf16*)(ws + 12845056);  // 4 MB  [B,N,CI]
  __bf16* gch = (__bf16*)(ws + 17039360);  // 4 MB  [B,CI,N]
  __bf16* yb  = (__bf16*)(ws + 21233664);  // 4 MB  [B,N,CI]

  k_cvt<<<128, 256, 0, stream>>>(wt, wtb, 32768);
  k_cvt<<<128, 256, 0, stream>>>(wp, wpb, 32768);
  k_cvt<<<128, 256, 0, stream>>>(wg, wgb, 32768);
  k_cvt<<<128, 256, 0, stream>>>(wo, wob, 32768);
  k_xpose<<<dim3(64, 4, 4), dim3(64, 4), 0, stream>>>(x, xt);
  k_proj<<<768, 128, 0, stream>>>(xt, wtb, wpb, wgb, bt, bp, bg, th, ph, gch);
  k_attn<<<256, 128, 0, stream>>>(th, ph, gch, yb);
  k_out<<<512, 128, 0, stream>>>(yb, wob, bo, gm, bb, mn, vr, x, out);
}
